// LinearTripleRCell_16801912062692
// MI455X (gfx1250) — compile-verified, matrix operand order checked by reading, not run on hardware
//
#include <hip/hip_runtime.h>
#include <hip/hip_bf16.h>

typedef __attribute__((ext_vector_type(4)))  float  v4f;
typedef __attribute__((ext_vector_type(8)))  float  v8f;
typedef __attribute__((ext_vector_type(8)))  __bf16 v8bf;
typedef __attribute__((ext_vector_type(16))) __bf16 v16bf;

namespace {
constexpr int T  = 1024;
constexpr int Bb = 32;
constexpr int D  = 1024;
constexpr int NWG  = 64;          // persistent workgroups; each owns D/NWG output columns
constexpr int COLS = D / NWG;     // 16 columns per WG
constexpr int LDW  = D + 8;       // padded LDS k-stride (bf16 elems): bank-conflict-free b128 reads
constexpr int SLICE = COLS * LDW; // one weight slice in LDS (bf16 elems)

constexpr size_t W_BYTES   = (size_t)4 * SLICE * sizeof(__bf16);       // 132096 B
constexpr size_t ACC_OFF   = W_BYTES;                                   // 8 waves * 16x16 f32
constexpr size_t HPREV_OFF = ACC_OFF + 8 * 256 * sizeof(float);
constexpr size_t BIAS_OFF  = HPREV_OFF + (size_t)Bb * COLS * sizeof(float);
constexpr size_t SMEM_BYTES = BIAS_OFF + 3 * COLS * sizeof(float);      // ~142.5 KB of 320 KB WGP LDS

constexpr size_t OUT_H_OFF = (size_t)T * Bb * D;  // offset of h history in d_out
}

// --------- WMMA fragment loaders (ISA 7.12.2 VGPR layouts, wave32) ----------
// A 16x32 bf16: lanes 0-15 -> row M=lane, K in {kb..kb+7, kb+16..kb+23} with kb=0;
//               lanes 16-31 -> same rows, kb=8. Two 16B loads per lane.
__device__ __forceinline__ v16bf frag_from_bf16(const __bf16* p) {
  v8bf lo = *(const v8bf*)(p);
  v8bf hi = *(const v8bf*)(p + 16);
  return __builtin_shufflevector(lo, hi, 0, 1, 2, 3, 4, 5, 6, 7,
                                 8, 9, 10, 11, 12, 13, 14, 15);
}

// Same fragment but converting fp32 source (x) -> bf16 on the fly.
__device__ __forceinline__ v16bf frag_from_f32(const float* p) {
  v4f f0 = *(const v4f*)(p);
  v4f f1 = *(const v4f*)(p + 4);
  v4f f2 = *(const v4f*)(p + 16);
  v4f f3 = *(const v4f*)(p + 20);
  v16bf r;
#pragma unroll
  for (int i = 0; i < 4; ++i) {
    r[i]      = (__bf16)f0[i];
    r[4 + i]  = (__bf16)f1[i];
    r[8 + i]  = (__bf16)f2[i];
    r[12 + i] = (__bf16)f3[i];
  }
  return r;
}

// Device-scope split barrier: one arrive counter + monotonic generation.
__device__ __forceinline__ void grid_barrier(unsigned* arrive, unsigned* gen) {
  __syncthreads();
  if (threadIdx.x == 0) {
    __threadfence();  // make this WG's global writes (hbuf, outputs) device-visible
    unsigned g = __hip_atomic_load(gen, __ATOMIC_ACQUIRE, __HIP_MEMORY_SCOPE_AGENT);
    unsigned prev =
        __hip_atomic_fetch_add(arrive, 1u, __ATOMIC_ACQ_REL, __HIP_MEMORY_SCOPE_AGENT);
    if (prev == NWG - 1) {
      __hip_atomic_store(arrive, 0u, __ATOMIC_RELAXED, __HIP_MEMORY_SCOPE_AGENT);
      __hip_atomic_fetch_add(gen, 1u, __ATOMIC_ACQ_REL, __HIP_MEMORY_SCOPE_AGENT);
    } else {
      while (__hip_atomic_load(gen, __ATOMIC_ACQUIRE, __HIP_MEMORY_SCOPE_AGENT) == g) {
        __builtin_amdgcn_s_sleep(2);
      }
    }
  }
  __syncthreads();
}

__global__ __launch_bounds__(256)
void rnn_scan_persistent(const float* __restrict__ x, const float* __restrict__ h0,
                         const float* __restrict__ R_h, const float* __restrict__ R_x,
                         const float* __restrict__ R_delta, const float* __restrict__ W_delta,
                         const float* __restrict__ b, const float* __restrict__ b_delta,
                         const float* __restrict__ b_gate, float* __restrict__ out,
                         unsigned* __restrict__ arrive, unsigned* __restrict__ gen,
                         __bf16* __restrict__ hbuf0, __bf16* __restrict__ hbuf1) {
  extern __shared__ char smem_raw[];
  __bf16* wlds = (__bf16*)smem_raw;                    // 4 weight slices, bf16, padded
  float*  accs = (float*)(smem_raw + ACC_OFF);         // 8 waves x 16x16 f32 tiles
  float*  hprev = (float*)(smem_raw + HPREV_OFF);      // this WG's fp32 h slice [32x16]
  float*  bias  = (float*)(smem_raw + BIAS_OFF);       // b / b_delta / b_gate slices

  const int tid  = threadIdx.x;
  const int wg   = blockIdx.x;
  const int col0 = wg * COLS;

  // ---- one-time init: stage weight slices fp32->bf16 into LDS ----
  for (int i = tid; i < 4 * COLS * D; i += 256) {
    int g   = i / (COLS * D);
    int rem = i - g * (COLS * D);
    int row = rem >> 10;        // local column index 0..15
    int k   = rem & (D - 1);
    const float* W = (g == 0) ? R_h : (g == 1) ? R_delta : (g == 2) ? R_x : W_delta;
    wlds[g * SLICE + row * LDW + k] = (__bf16)W[(size_t)(col0 + row) * D + k];
  }
  if (tid < 3 * COLS) {
    int which = tid / COLS, n = tid % COLS;
    const float* bp = (which == 0) ? b : (which == 1) ? b_delta : b_gate;
    bias[tid] = bp[col0 + n];
  }
  // h[0] = h0 (fp32 history output, fp32 LDS copy, bf16 exchange buffer)
  for (int e = tid; e < Bb * COLS; e += 256) {
    int m = e >> 4, n = e & 15;
    int gc = col0 + n;
    float hv = h0[(size_t)m * D + gc];
    out[OUT_H_OFF + (size_t)m * D + gc] = hv;
    hbuf0[m * D + gc] = (__bf16)hv;
    hprev[e] = hv;
  }
  grid_barrier(arrive, gen);

  const int lane  = tid & 31;
  const int wave  = tid >> 5;
  const int gemm  = wave >> 1;  // 0:R_h 1:R_delta 2:R_x 3:W_delta
  const int mt    = wave & 1;   // batch-row tile (0..15 / 16..31)
  const int ra    = mt * 16 + (lane & 15);
  const int kboff = (lane >> 4) << 3;
  const __bf16* wb = wlds + gemm * SLICE + (lane & 15) * LDW;  // B frag: lane -> N

  for (int t = 0; t < T; ++t) {
    const __bf16* hcur = (t & 1) ? hbuf1 : hbuf0;
    __bf16*       hnxt = (t & 1) ? hbuf0 : hbuf1;

    // ---- per-wave 16x16 output tile, K = 1024 in steps of 32 ----
    v8f acc;
#pragma unroll
    for (int i = 0; i < 8; ++i) acc[i] = 0.f;

    if (gemm < 2) {  // h_prev @ R^T : A from bf16 exchange buffer (L2 resident)
      const __bf16* ha = hcur + ra * D;
#pragma unroll 4
      for (int kk = 0; kk < D; kk += 32) {
        v16bf afr = frag_from_bf16(ha + kk + kboff);
        v16bf bfr = frag_from_bf16(wb + kk + kboff);   // ds_load_b128 from LDS
        acc = __builtin_amdgcn_wmma_f32_16x16x32_bf16(false, afr, false, bfr,
                                                      (short)0, acc, false, false);
      }
    } else {         // x_t @ R^T : A converted fp32->bf16 from L2-resident x
      const float* xa = x + ((size_t)t * Bb + ra) * D;
#pragma unroll 4
      for (int kk = 0; kk < D; kk += 32) {
        v16bf afr = frag_from_f32(xa + kk + kboff);
        v16bf bfr = frag_from_bf16(wb + kk + kboff);
        acc = __builtin_amdgcn_wmma_f32_16x16x32_bf16(false, afr, false, bfr,
                                                      (short)0, acc, false, false);
      }
      if (t + 1 < T) __builtin_prefetch(xa + (size_t)Bb * D, 0, 1);  // next step's rows
    }

    // C/D layout: VGPR r -> M = r + 8*(lane>=16), N = lane&15
    {
      float* at = accs + wave * 256;
      int n  = lane & 15;
      int m0 = (lane >> 4) << 3;
#pragma unroll
      for (int r = 0; r < 8; ++r) at[(m0 + r) * 16 + n] = acc[r];
    }
    __syncthreads();

    // ---- elementwise recurrence on this WG's [32 x 16] slice ----
    for (int e = tid; e < Bb * COLS; e += 256) {
      int m = e >> 4, n = e & 15;
      int gc = col0 + n;
      int tl = m >> 4, ml = m & 15;
      float hRh = accs[(0 * 2 + tl) * 256 + ml * 16 + n];
      float hRd = accs[(1 * 2 + tl) * 256 + ml * 16 + n];
      float xRx = accs[(2 * 2 + tl) * 256 + ml * 16 + n];
      float xWd = accs[(3 * 2 + tl) * 256 + ml * 16 + n];
      float hp  = hprev[e];
      float v   = xRx + bias[n] + hRh;
      float u   = xWd + bias[16 + n] + hRd;
      float dl  = 1.f / (1.f + __expf(-u));
      float hn  = (1.f - dl) * hp + dl * tanhf(v);
      float xg  = x[((size_t)t * Bb + m) * D + gc];
      float z   = hn + xg + bias[32 + n];
      float sil = z / (1.f + __expf(-z));
      out[((size_t)t * Bb + m) * D + gc]                   = hn * sil;   // outputs[t]
      out[OUT_H_OFF + ((size_t)(t + 1) * Bb + m) * D + gc] = hn;         // h[t+1]
      hnxt[m * D + gc] = (__bf16)hn;                                     // bf16 exchange
      hprev[e] = hn;
    }
    grid_barrier(arrive, gen);  // h_new visible device-wide before next step
  }
}

extern "C" void kernel_launch(void* const* d_in, const int* in_sizes, int n_in,
                              void* d_out, int out_size, void* d_ws, size_t ws_size,
                              hipStream_t stream) {
  (void)in_sizes; (void)n_in; (void)out_size; (void)ws_size;
  const float* x       = (const float*)d_in[0];
  const float* h0      = (const float*)d_in[1];
  const float* R_h     = (const float*)d_in[2];
  const float* R_x     = (const float*)d_in[3];
  const float* R_delta = (const float*)d_in[4];
  const float* W_delta = (const float*)d_in[5];
  const float* b       = (const float*)d_in[6];
  const float* b_delta = (const float*)d_in[7];
  const float* b_gate  = (const float*)d_in[8];
  float* out = (float*)d_out;

  // workspace: [0,256) barrier counters; then two bf16 h ping-pong buffers (64 KB each)
  unsigned* arrive = (unsigned*)d_ws;
  unsigned* gen    = arrive + 1;
  __bf16* hbuf0 = (__bf16*)((char*)d_ws + 256);
  __bf16* hbuf1 = hbuf0 + (size_t)Bb * D;

  hipMemsetAsync(d_ws, 0, 256, stream);  // reset barrier state every call (capture-safe)

  (void)hipFuncSetAttribute(reinterpret_cast<const void*>(&rnn_scan_persistent),
                            hipFuncAttributeMaxDynamicSharedMemorySize, (int)SMEM_BYTES);

  rnn_scan_persistent<<<NWG, 256, SMEM_BYTES, stream>>>(
      x, h0, R_h, R_x, R_delta, W_delta, b, b_delta, b_gate, out,
      arrive, gen, hbuf0, hbuf1);
}
